// DIR_82446192214563
// MI455X (gfx1250) — matrix operand order repaired and checked by reading, not compiled, not run on hardware
//
#include <hip/hip_runtime.h>
#include <hip/hip_bf16.h>

// ---------------------------------------------------------------------------
// Problem constants (from the reference)
// ---------------------------------------------------------------------------
constexpr int G    = 128;
constexpr int NPG  = 256;
constexpr int EPG  = 4096;
constexpr int Nn   = G * NPG;     // 32768 nodes
constexpr int En   = G * EPG;     // 524288 edges
constexpr int Hh   = 64;          // hidden dim
constexpr int Kk   = EPG / 2;     // 2048 kept edges per graph
constexpr int GKt  = G * Kk;      // 262144

typedef __attribute__((ext_vector_type(16))) _Float16 v16h;
typedef __attribute__((ext_vector_type(8)))  _Float16 v8h;
typedef __attribute__((ext_vector_type(8)))  float    v8f;

// Load 8 consecutive f32 from p, convert to f16 into fragment halves [off..off+7]
__device__ __forceinline__ void ld8h(v16h& d, int off, const float* p) {
  const float4 u = *reinterpret_cast<const float4*>(p);
  const float4 v = *reinterpret_cast<const float4*>(p + 4);
  d[off + 0] = (_Float16)u.x; d[off + 1] = (_Float16)u.y;
  d[off + 2] = (_Float16)u.z; d[off + 3] = (_Float16)u.w;
  d[off + 4] = (_Float16)v.x; d[off + 5] = (_Float16)v.y;
  d[off + 6] = (_Float16)v.z; d[off + 7] = (_Float16)v.w;
}

// Merge two 8-half runs (16B aligned loads) into one A/B fragment
__device__ __forceinline__ v16h merge8(const _Float16* p0, const _Float16* p1) {
  const v8h lo = *reinterpret_cast<const v8h*>(p0);
  const v8h hi = *reinterpret_cast<const v8h*>(p1);
  return __builtin_shufflevector(lo, hi, 0, 1, 2, 3, 4, 5, 6, 7,
                                 8, 9, 10, 11, 12, 13, 14, 15);
}

// ---------------------------------------------------------------------------
// Kernel: per-layer projections  A = X@W1^T+b1 ; B = X@W2^T ; C = X@W3^T+b3
// X: (Nn,64). One block = 4 waves handles one 16-row tile; each wave does
// 3 of the 12 column tiles (3 outputs x 4 tiles of 16). K=64 -> 2 WMMA steps.
// A-frag layout (16-bit A 16x32, ISA 7.12.2): lane<16: M=lane, K in {0..7,16..23};
// lane>=16: M=lane-16, K in {8..15,24..31}.  B[k][n] = W[n][k] -> per-lane two
// 8-float runs of row n of W, in the same K order as A.
// ---------------------------------------------------------------------------
__global__ __launch_bounds__(128) void dirgnn_proj(
    const float* __restrict__ X,
    const float* __restrict__ W1, const float* __restrict__ b1,
    const float* __restrict__ W2, const float* __restrict__ W3,
    const float* __restrict__ b3,
    float* __restrict__ OA, float* __restrict__ OB, float* __restrict__ OC) {
  const int row0 = blockIdx.x * 16;
  const int lane = threadIdx.x & 31;
  const int wv   = threadIdx.x >> 5;
  const int m    = row0 + (lane & 15);

  v16h af[2];
#pragma unroll
  for (int s = 0; s < 2; ++s) {
    const int kb = 32 * s;
    const int k0 = kb + ((lane < 16) ? 0 : 8);
    const int k1 = kb + ((lane < 16) ? 16 : 24);
    ld8h(af[s], 0, X + m * Hh + k0);
    ld8h(af[s], 8, X + m * Hh + k1);
  }

#pragma unroll
  for (int t3 = 0; t3 < 3; ++t3) {
    const int t     = wv * 3 + t3;   // 0..11
    const int which = t >> 2;        // 0 -> A, 1 -> B, 2 -> C
    const int nc0   = (t & 3) * 16;
    const float* W    = (which == 0) ? W1 : ((which == 1) ? W2 : W3);
    const float* bias = (which == 0) ? b1 : ((which == 2) ? b3 : nullptr);
    float*       O    = (which == 0) ? OA : ((which == 1) ? OB : OC);
    const int n = nc0 + (lane & 15);

    v8f acc = {};
#pragma unroll
    for (int s = 0; s < 2; ++s) {
      const int kb = 32 * s + ((lane < 16) ? 0 : 8);
      v16h bf;
      ld8h(bf, 0, W + n * Hh + kb);
      ld8h(bf, 8, W + n * Hh + kb + 16);
      acc = __builtin_amdgcn_wmma_f32_16x16x32_f16(
          false, af[s], false, bf, (short)0, acc, false, false);
    }
    const float bv = bias ? bias[n] : 0.f;
#pragma unroll
    for (int r = 0; r < 8; ++r) {
      const int mr = row0 + ((lane < 16) ? r : r + 8);
      O[mr * Hh + n] = acc[r] + bv;   // C/D layout: lane<16 -> M=r, N=lane
    }
  }
}

// ---------------------------------------------------------------------------
// Kernel: segment-sum scatter  AGG[col] += ew*(A[row]-B[col])
// 16 threads per edge, 4 features each -> native global f32 atomics.
// ---------------------------------------------------------------------------
__global__ void dirgnn_scatter(
    const float* __restrict__ A, const float* __restrict__ B,
    const float* __restrict__ ew,
    const int* __restrict__ row, const int* __restrict__ col,
    float* __restrict__ AGG) {
  const long long idx = (long long)blockIdx.x * 256 + threadIdx.x;
  const int e = (int)(idx >> 4);
  const int f = (int)(idx & 15) * 4;
  if (e >= En) return;
  const int r = row[e], c = col[e];
  const float w = ew[e];
  const float4 a = *reinterpret_cast<const float4*>(A + (long long)r * Hh + f);
  const float4 b = *reinterpret_cast<const float4*>(B + (long long)c * Hh + f);
  float* dst = AGG + (long long)c * Hh + f;
  unsafeAtomicAdd(dst + 0, w * (a.x - b.x));
  unsafeAtomicAdd(dst + 1, w * (a.y - b.y));
  unsafeAtomicAdd(dst + 2, w * (a.z - b.z));
  unsafeAtomicAdd(dst + 3, w * (a.w - b.w));
}

__global__ void dirgnn_relu(float* __restrict__ h) {
  const long long i = (long long)blockIdx.x * 256 + threadIdx.x;
  if (i < (long long)Nn * Hh) h[i] = fmaxf(h[i], 0.f);
}

// Elementwise f32 -> f16 pre-conversion (amortizes cvt out of the WMMA loop
// and halves the L2 gather traffic of the edge MLP).
__global__ void dirgnn_cvt_f16(const float* __restrict__ src,
                               _Float16* __restrict__ dst, long long n) {
  const long long i = (long long)blockIdx.x * 256 + threadIdx.x;
  if (i < n) dst[i] = (_Float16)src[i];
}

// ---------------------------------------------------------------------------
// Kernel: edge MLP + score.
// Block = 512 threads = 16 waves, handles 32 edges (2 M-tiles of 16, reusing
// B fragments). Wave w owns output columns [16w,16w+16). K=128 -> 4 WMMA.
// A[m][k] = k<64 ? H[row[e]][k] : H[col[e]][k-64]  (8-runs never cross 64).
// H and Wm1 are pre-converted f16: fragments are pure 16B/32B loads, no cvt.
// Score reduction: relu(y+bm1)*Wm2 summed across 256 cols via ds_add_f32.
// ---------------------------------------------------------------------------
__global__ __launch_bounds__(512) void dirgnn_edge_mlp(
    const _Float16* __restrict__ Hf,
    const int* __restrict__ row, const int* __restrict__ col,
    const _Float16* __restrict__ Wm1f, const float* __restrict__ bm1,
    const float* __restrict__ Wm2, const float* __restrict__ bm2,
    float* __restrict__ score) {
  __shared__ float lds_s[32];
  const int lane = threadIdx.x & 31;
  const int wv   = threadIdx.x >> 5;
  const int e0   = blockIdx.x * 32;
  const int n    = lane & 15;
  const int ng   = wv * 16 + n;    // 0..255

  v16h bf[4];
#pragma unroll
  for (int s = 0; s < 4; ++s) {
    const int kb = 32 * s + ((lane < 16) ? 0 : 16);
    bf[s] = *reinterpret_cast<const v16h*>(Wm1f + ng * 128 + kb);  // 32B load
  }
  const float bm = bm1[ng];
  const float wn = Wm2[ng];

  if (threadIdx.x < 32) lds_s[threadIdx.x] = 0.f;
  __syncthreads();

#pragma unroll
  for (int mt = 0; mt < 2; ++mt) {
    const int e = e0 + mt * 16 + n;
    const _Float16* hr = Hf + (long long)row[e] * Hh;
    const _Float16* hc = Hf + (long long)col[e] * Hh;
    v8f acc = {};
#pragma unroll
    for (int s = 0; s < 4; ++s) {
      const int k0 = 32 * s + ((lane < 16) ? 0 : 16);
      const int k1 = k0 + 8;
      const _Float16* p0 = (k0 < 64) ? hr + k0 : hc + (k0 - 64);
      const _Float16* p1 = (k1 < 64) ? hr + k1 : hc + (k1 - 64);
      const v16h af = merge8(p0, p1);
      acc = __builtin_amdgcn_wmma_f32_16x16x32_f16(
          false, af, false, bf[s], (short)0, acc, false, false);
    }
#pragma unroll
    for (int r8 = 0; r8 < 8; ++r8) {
      float y = acc[r8] + bm;
      y = fmaxf(y, 0.f) * wn;
      const int mr = mt * 16 + ((lane < 16) ? r8 : r8 + 8);
      atomicAdd(&lds_s[mr], y);   // ds_add_f32
    }
  }
  __syncthreads();
  if (threadIdx.x < 32) score[e0 + threadIdx.x] = lds_s[threadIdx.x] + bm2[0];
}

// ---------------------------------------------------------------------------
// Kernel: per-graph descending bitonic argsort of 4096 scores in LDS (32KB),
// then emit causal_w / conf_w and global kept/dropped edge ids.
// ---------------------------------------------------------------------------
__global__ __launch_bounds__(512) void dirgnn_sort(
    const float* __restrict__ score,
    float* __restrict__ causal_w, float* __restrict__ conf_w,
    int* __restrict__ kg, int* __restrict__ dg) {
  __shared__ float ss[EPG];
  __shared__ int   si[EPG];
  const int g = blockIdx.x;
  const float* s = score + (long long)g * EPG;
  for (int i = threadIdx.x; i < EPG; i += 512) { ss[i] = s[i]; si[i] = i; }
  __syncthreads();
  for (int k = 2; k <= EPG; k <<= 1) {
    for (int j = k >> 1; j > 0; j >>= 1) {
      for (int i = threadIdx.x; i < EPG; i += 512) {
        const int ixj = i ^ j;
        if (ixj > i) {
          const bool up = (i & k) == 0;
          const float a = ss[i], b = ss[ixj];
          const bool sw = up ? (a < b) : (a > b);   // descending final order
          if (sw) {
            ss[i] = b; ss[ixj] = a;
            const int t = si[i]; si[i] = si[ixj]; si[ixj] = t;
          }
        }
      }
      __syncthreads();
    }
  }
  for (int i = threadIdx.x; i < Kk; i += 512) {
    causal_w[g * Kk + i] = ss[i];
    kg[g * Kk + i]       = si[i] + g * EPG;
    conf_w[g * Kk + i]   = -ss[Kk + i];
    dg[g * Kk + i]       = si[Kk + i] + g * EPG;
  }
}

// ---------------------------------------------------------------------------
// Relabel pipeline: zero masks, mark present, block-wide scan (N=32768 =
// 1024 threads x 32 elems), remap edge endpoints.
// ---------------------------------------------------------------------------
__global__ void dirgnn_zero(int* __restrict__ pc, int* __restrict__ pf) {
  const int i = blockIdx.x * 256 + threadIdx.x;
  if (i < Nn) { pc[i] = 0; pf[i] = 0; }
}

__global__ void dirgnn_mark(
    const int* __restrict__ kg, const int* __restrict__ dg,
    const int* __restrict__ row, const int* __restrict__ col,
    int* __restrict__ pc, int* __restrict__ pf) {
  const int j = blockIdx.x * 256 + threadIdx.x;
  if (j < GKt) {
    const int e = kg[j]; pc[row[e]] = 1; pc[col[e]] = 1;
  } else if (j < 2 * GKt) {
    const int e = dg[j - GKt]; pf[row[e]] = 1; pf[col[e]] = 1;
  }
}

__global__ __launch_bounds__(1024) void dirgnn_scan(
    const int* __restrict__ pc, const int* __restrict__ pf,
    int* __restrict__ nidc, int* __restrict__ nidf,
    float* __restrict__ maskc, float* __restrict__ maskf) {
  __shared__ int lds[1024];
  const int* p  = (blockIdx.x == 0) ? pc : pf;
  int* nid      = (blockIdx.x == 0) ? nidc : nidf;
  float* msk    = (blockIdx.x == 0) ? maskc : maskf;
  const int t = threadIdx.x;
  const int base = t * 32;
  int sum = 0;
  for (int q = 0; q < 32; ++q) sum += p[base + q];
  lds[t] = sum;
  __syncthreads();
  for (int off = 1; off < 1024; off <<= 1) {
    const int v = (t >= off) ? lds[t - off] : 0;
    __syncthreads();
    lds[t] += v;
    __syncthreads();
  }
  int run = (t == 0) ? 0 : lds[t - 1];
  for (int q = 0; q < 32; ++q) {
    const int pv = p[base + q];
    run += pv;
    nid[base + q] = pv ? run - 1 : -1;
    msk[base + q] = (float)pv;
  }
}

__global__ void dirgnn_map(
    const int* __restrict__ kg, const int* __restrict__ dg,
    const int* __restrict__ row, const int* __restrict__ col,
    const int* __restrict__ nidc, const int* __restrict__ nidf,
    float* __restrict__ cei, float* __restrict__ fei) {
  const int j = blockIdx.x * 256 + threadIdx.x;
  if (j < GKt) {
    const int e = kg[j];
    cei[j]       = (float)nidc[row[e]];
    cei[GKt + j] = (float)nidc[col[e]];
  } else if (j < 2 * GKt) {
    const int jj = j - GKt;
    const int e = dg[jj];
    fei[jj]       = (float)nidf[row[e]];
    fei[GKt + jj] = (float)nidf[col[e]];
  }
}

// ---------------------------------------------------------------------------
// Host launcher
// ---------------------------------------------------------------------------
extern "C" void kernel_launch(void* const* d_in, const int* in_sizes, int n_in,
                              void* d_out, int out_size, void* d_ws, size_t ws_size,
                              hipStream_t stream) {
  const float* x   = (const float*)d_in[0];
  const float* ew  = (const float*)d_in[1];
  const float* W11 = (const float*)d_in[2];
  const float* b11 = (const float*)d_in[3];
  const float* W12 = (const float*)d_in[4];
  const float* W13 = (const float*)d_in[5];
  const float* b13 = (const float*)d_in[6];
  const float* W21 = (const float*)d_in[7];
  const float* b21 = (const float*)d_in[8];
  const float* W22 = (const float*)d_in[9];
  const float* W23 = (const float*)d_in[10];
  const float* b23 = (const float*)d_in[11];
  const float* Wm1 = (const float*)d_in[12];
  const float* bm1 = (const float*)d_in[13];
  const float* Wm2 = (const float*)d_in[14];
  const float* bm2 = (const float*)d_in[15];
  const int*   ei  = (const int*)d_in[16];   // (2,E) int32 (default JAX x32)
  const int* row = ei;
  const int* col = ei + En;

  // Workspace layout (bytes)
  char* ws = (char*)d_ws;
  float*    wsA   = (float*)(ws);                              // N*64 f32
  float*    wsB   = (float*)(ws + 8388608);                    // N*64
  float*    h1    = (float*)(ws + 16777216);                   // N*64 (AGG1 -> h1)
  float*    h2    = (float*)(ws + 25165824);                   // N*64 (AGG2 -> h2)
  int*      kg    = (int*)  (ws + 33554432);                   // GKt
  int*      dg    = (int*)  (ws + 34603008);                   // GKt
  int*      pc    = (int*)  (ws + 35651584);                   // N
  int*      pf    = (int*)  (ws + 35651584 + 131072);          // N
  int*      nidc  = (int*)  (ws + 35651584 + 2 * 131072);      // N
  int*      nidf  = (int*)  (ws + 35651584 + 3 * 131072);      // N
  _Float16* h2f   = (_Float16*)(ws + 36175872);                // N*64 f16 (4MB)
  _Float16* wm1f  = (_Float16*)(ws + 40370176);                // 256*128 f16

  // Output layout (all as float per harness convention)
  float* out      = (float*)d_out;
  float* score    = out;                         // E
  float* causal_w = out + En;                    // GKt
  float* conf_w   = out + En + GKt;              // GKt
  float* cei      = out + En + 2 * GKt;          // 2*GKt
  float* fei      = cei + 2 * GKt;               // 2*GKt
  float* maskc    = fei + 2 * GKt;               // N
  float* maskf    = maskc + Nn;                  // N

  dirgnn_zero<<<Nn / 256, 256, 0, stream>>>(pc, pf);

  // Layer 1: proj -> scatter -> relu  (h1 = relu(AGG + C); C pre-stored in h1)
  dirgnn_proj<<<Nn / 16, 128, 0, stream>>>(x, W11, b11, W12, W13, b13,
                                           wsA, wsB, h1);
  dirgnn_scatter<<<(En * 16) / 256, 256, 0, stream>>>(wsA, wsB, ew, row, col, h1);
  dirgnn_relu<<<(Nn * Hh) / 256, 256, 0, stream>>>(h1);

  // Layer 2
  dirgnn_proj<<<Nn / 16, 128, 0, stream>>>(h1, W21, b21, W22, W23, b23,
                                           wsA, wsB, h2);
  dirgnn_scatter<<<(En * 16) / 256, 256, 0, stream>>>(wsA, wsB, ew, row, col, h2);

  // Pre-convert h2 and Wm1 to f16 for the WMMA edge MLP
  dirgnn_cvt_f16<<<(Nn * Hh) / 256, 256, 0, stream>>>(h2, h2f,
                                                      (long long)Nn * Hh);
  dirgnn_cvt_f16<<<(256 * 128) / 256, 256, 0, stream>>>(Wm1, wm1f,
                                                        (long long)256 * 128);

  // Edge MLP + scores
  dirgnn_edge_mlp<<<En / 32, 512, 0, stream>>>(h2f, row, col, wm1f, bm1, Wm2,
                                               bm2, score);

  // Per-graph top-K split
  dirgnn_sort<<<G, 512, 0, stream>>>(score, causal_w, conf_w, kg, dg);

  // Relabel
  dirgnn_mark<<<(2 * GKt) / 256, 256, 0, stream>>>(kg, dg, row, col, pc, pf);
  dirgnn_scan<<<2, 1024, 0, stream>>>(pc, pf, nidc, nidf, maskc, maskf);
  dirgnn_map<<<(2 * GKt) / 256, 256, 0, stream>>>(kg, dg, row, col, nidc, nidf,
                                                  cei, fei);
}
